// Model_34144990003485
// MI455X (gfx1250) — matrix operand order repaired and checked by reading, not compiled, not run on hardware
//
#include <hip/hip_runtime.h>
#include <hip/hip_bf16.h>
#include <stdint.h>

// ---------------------------------------------------------------------------
// Problem constants (from reference)
// ---------------------------------------------------------------------------
#define Hd      512
#define Ed      321
#define Cd      321
#define Td      96
#define Pd      96
#define Bd      512
#define G4H     2048          // 4*H
#define KPAD_E  352           // ceil(321/32)*32
#define KT_E    11            // 352/32
#define KT_H    16            // 512/32
#define MT2     16            // 512/32  (32-row macro tiles)
#define NT2     64            // 2048/32 (32-col macro tiles)
#define MT_B    32            // 512/16
#define NT_FC   21            // ceil(321/16) -> 336/16

typedef __attribute__((ext_vector_type(16))) __bf16 v16bf;
typedef __attribute__((ext_vector_type(8)))  float  v8f;

// ---------------------------------------------------------------------------
// Helpers
// ---------------------------------------------------------------------------
__device__ __forceinline__ unsigned short f2bf(float f) {
  union { float f; unsigned u; } x; x.f = f;
  unsigned u = x.u;
  u = u + 0x7FFFu + ((u >> 16) & 1u);     // round-to-nearest-even
  return (unsigned short)(u >> 16);
}

__device__ __forceinline__ v16bf load_frag_pair(const unsigned short* p0,
                                                const unsigned short* p1) {
  union { uint4 q[2]; v16bf v; } u;
  u.q[0] = *(const uint4*)p0;   // 16B aligned by construction
  u.q[1] = *(const uint4*)p1;
  return u.v;
}

__device__ __forceinline__ float sigf(float x) {
  return 1.0f / (1.0f + __expf(-x));
}

#define WMMA_BF16(A, B, C) \
  __builtin_amdgcn_wmma_f32_16x16x32_bf16(false, (A), false, (B), (short)0, (C), false, false)

// ---------------------------------------------------------------------------
// Weight packing: fp32 (Nw x Kw, row-major) -> bf16 WMMA B-fragment layout.
// Layout: [tN][tK][lane(32)][16 halfs]; element (lane,j):
//   n = tN*16 + (lane & 15)
//   k = tK*32 + ((lane>>4)<<3) + (j&7) + ((j>>3)<<4)  (lanes 0-15: K 0..7,16..23; 16-31: K 8..15,24..31)
//   value = W[n][k]  (B matrix = W^T so B[k][n] = W[n][k]), zero-padded OOB.
// ---------------------------------------------------------------------------
__global__ void pack_w_kernel(const float* __restrict__ W,
                              unsigned short* __restrict__ Wp,
                              int Nw, int Kw, int kTiles, int total) {
  int idx = blockIdx.x * blockDim.x + threadIdx.x;
  if (idx >= total) return;
  int frag = idx >> 9;          // 512 halfs per (tN,tK) fragment
  int rem  = idx & 511;
  int lane = rem >> 4;
  int j    = rem & 15;
  int tN   = frag / kTiles;
  int tK   = frag % kTiles;
  int n = tN * 16 + (lane & 15);
  int k = tK * 32 + ((lane >> 4) << 3) + (j & 7) + ((j >> 3) << 4);
  float v = (n < Nw && k < Kw) ? W[(size_t)n * Kw + k] : 0.0f;
  Wp[idx] = f2bf(v);
}

__global__ void bias_sum_kernel(const float* __restrict__ a,
                                const float* __restrict__ b,
                                float* __restrict__ o) {
  int i = blockIdx.x * blockDim.x + threadIdx.x;
  if (i < G4H) o[i] = a[i] + b[i];
}

// Convert one encoder timestep x_enc[:, t, :] -> bf16, zero-padded to KPAD_E.
__global__ void conv_x_kernel(const float* __restrict__ x,
                              unsigned short* __restrict__ Xs, int t) {
  int idx = blockIdx.x * blockDim.x + threadIdx.x;
  if (idx >= Bd * KPAD_E) return;
  int b = idx / KPAD_E;
  int k = idx - b * KPAD_E;
  float v = (k < Ed) ? x[((size_t)b * Td + t) * Ed + k] : 0.0f;
  Xs[idx] = f2bf(v);
}

// ---------------------------------------------------------------------------
// Fused gates GEMM:  gates(B x 4H, f32) = Xa @ WpA^T + Xh @ WpH^T + bias
// One wave per 32x32 macro tile (2x2 subtiles of 16x16): each K-tile issues
// 8 b128 loads feeding 4 v_wmma_f32_16x16x32_bf16 (A/B fragments reused 2x).
// ---------------------------------------------------------------------------
__global__ __launch_bounds__(256) void gemm_gates_kernel(
    const unsigned short* __restrict__ Xa, int kTilesA,
    const unsigned short* __restrict__ WpA,
    const unsigned short* __restrict__ Xh, int kTilesH,
    const unsigned short* __restrict__ WpH,
    const float* __restrict__ bias,
    float* __restrict__ out) {
  const int lane = threadIdx.x & 31;
  const int wave = threadIdx.x >> 5;
  const int tile = blockIdx.x * 8 + wave;
  if (tile >= MT2 * NT2) return;             // wave-uniform: EXEC all-1 for WMMA
  const int tM = tile / NT2;                 // 32-row macro tile
  const int tN = tile % NT2;                 // 32-col macro tile
  const int kOff = (lane >> 4) << 3;         // 0 or 8 (A-fragment K split)

  v8f acc00 = {0.f,0.f,0.f,0.f,0.f,0.f,0.f,0.f};
  v8f acc01 = acc00, acc10 = acc00, acc11 = acc00;

  {   // input (x) contribution
    const int Kpad = kTilesA * 32;
    const unsigned short* Xrow0 = Xa + (size_t)(tM * 32 + (lane & 15)) * Kpad;
    const unsigned short* Xrow1 = Xrow0 + (size_t)16 * Kpad;
    const unsigned short* Wn0 = WpA + ((size_t)(tN * 2) * kTilesA) * 512 + lane * 16;
    const unsigned short* Wn1 = Wn0 + (size_t)kTilesA * 512;
    for (int tk = 0; tk < kTilesA; ++tk) {
      const unsigned short* xp0 = Xrow0 + tk * 32 + kOff;
      const unsigned short* xp1 = Xrow1 + tk * 32 + kOff;
      v16bf a0 = load_frag_pair(xp0, xp0 + 16);
      v16bf a1 = load_frag_pair(xp1, xp1 + 16);
      v16bf b0 = load_frag_pair(Wn0, Wn0 + 8);
      v16bf b1 = load_frag_pair(Wn1, Wn1 + 8);
      Wn0 += 512; Wn1 += 512;
      acc00 = WMMA_BF16(a0, b0, acc00);
      acc01 = WMMA_BF16(a0, b1, acc01);
      acc10 = WMMA_BF16(a1, b0, acc10);
      acc11 = WMMA_BF16(a1, b1, acc11);
    }
  }
  {   // recurrent (h) contribution
    const int Kpad = kTilesH * 32;
    const unsigned short* Xrow0 = Xh + (size_t)(tM * 32 + (lane & 15)) * Kpad;
    const unsigned short* Xrow1 = Xrow0 + (size_t)16 * Kpad;
    const unsigned short* Wn0 = WpH + ((size_t)(tN * 2) * kTilesH) * 512 + lane * 16;
    const unsigned short* Wn1 = Wn0 + (size_t)kTilesH * 512;
    for (int tk = 0; tk < kTilesH; ++tk) {
      const unsigned short* xp0 = Xrow0 + tk * 32 + kOff;
      const unsigned short* xp1 = Xrow1 + tk * 32 + kOff;
      v16bf a0 = load_frag_pair(xp0, xp0 + 16);
      v16bf a1 = load_frag_pair(xp1, xp1 + 16);
      v16bf b0 = load_frag_pair(Wn0, Wn0 + 8);
      v16bf b1 = load_frag_pair(Wn1, Wn1 + 8);
      Wn0 += 512; Wn1 += 512;
      acc00 = WMMA_BF16(a0, b0, acc00);
      acc01 = WMMA_BF16(a0, b1, acc01);
      acc10 = WMMA_BF16(a1, b0, acc10);
      acc11 = WMMA_BF16(a1, b1, acc11);
    }
  }

  // C/D layout: VGPR r, lane L -> row M = r + 8*(L>=16), col N = L&15
  const int n0 = tN * 32 + (lane & 15);
  const float bv0 = bias[n0];
  const float bv1 = bias[n0 + 16];
  const int mb = tM * 32 + ((lane >> 4) << 3);
  float* orow0 = out + (size_t)mb * G4H + n0;          // rows mb..mb+7
  float* orow1 = orow0 + (size_t)16 * G4H;             // rows mb+16..mb+23
#pragma unroll
  for (int r = 0; r < 8; ++r) {
    orow0[(size_t)r * G4H]      = acc00[r] + bv0;
    orow0[(size_t)r * G4H + 16] = acc01[r] + bv1;
    orow1[(size_t)r * G4H]      = acc10[r] + bv0;
    orow1[(size_t)r * G4H + 16] = acc11[r] + bv1;
  }
}

// ---------------------------------------------------------------------------
// LSTM cell pointwise: gate order i, f, g, o (reference split order)
// c (fp32) updated in place, h written as bf16 for the next GEMM.
// ---------------------------------------------------------------------------
__global__ void cell_kernel(const float* __restrict__ gates,
                            float* __restrict__ c,
                            unsigned short* __restrict__ hbf) {
  int idx = blockIdx.x * blockDim.x + threadIdx.x;  // B*H = 262144 exact
  int b = idx >> 9;
  int n = idx & 511;
  const float* g = gates + (size_t)b * G4H;
  float gi = sigf(g[n]);
  float gf = sigf(g[Hd + n]);
  float gg = tanhf(g[2 * Hd + n]);
  float go = sigf(g[3 * Hd + n]);
  float cv = gf * c[idx] + gi * gg;
  c[idx] = cv;
  hbf[idx] = f2bf(go * tanhf(cv));
}

// ---------------------------------------------------------------------------
// Decoder FC: out = h1 @ fc_W^T + fc_b ; writes fp32 to d_out[:, t, :] and
// feeds bf16 back into the padded decoder input buffer.
// ---------------------------------------------------------------------------
__global__ __launch_bounds__(256) void fc_kernel(
    const unsigned short* __restrict__ Xh,     // h1 bf16 (B x 512)
    const unsigned short* __restrict__ Wp,     // packed fc_W (NT_FC x KT_H frags)
    const float* __restrict__ fcb,             // raw fc_b (321)
    float* __restrict__ dout,                  // (B, P, C) fp32
    unsigned short* __restrict__ inp,          // (B, KPAD_E) bf16 (pad stays 0)
    int t) {
  const int lane = threadIdx.x & 31;
  const int wave = threadIdx.x >> 5;
  const int tile = blockIdx.x * 8 + wave;
  if (tile >= MT_B * NT_FC) return;            // wave-uniform
  const int tM = tile / NT_FC;
  const int tN = tile % NT_FC;
  const int kOff = (lane >> 4) << 3;

  v8f acc = {0.f, 0.f, 0.f, 0.f, 0.f, 0.f, 0.f, 0.f};
  const unsigned short* Xrow = Xh + (size_t)(tM * 16 + (lane & 15)) * 512;
  const unsigned short* Wn = Wp + ((size_t)tN * KT_H) * 512 + lane * 16;
  for (int tk = 0; tk < KT_H; ++tk) {
    const unsigned short* xp = Xrow + tk * 32 + kOff;
    v16bf a = load_frag_pair(xp, xp + 16);
    v16bf b = load_frag_pair(Wn, Wn + 8);
    Wn += 512;
    acc = WMMA_BF16(a, b, acc);
  }

  const int n = tN * 16 + (lane & 15);
  if (n < Cd) {                                // divergence after WMMA is fine
    const float bv = fcb[n];
    const int mbase = tM * 16 + ((lane >> 4) << 3);
#pragma unroll
    for (int r = 0; r < 8; ++r) {
      int m = mbase + r;
      float v = acc[r] + bv;
      dout[((size_t)m * Pd + t) * Cd + n] = v;
      inp[(size_t)m * KPAD_E + n] = f2bf(v);
    }
  }
}

// ---------------------------------------------------------------------------
// Host-side orchestration
// ---------------------------------------------------------------------------
extern "C" void kernel_launch(void* const* d_in, const int* in_sizes, int n_in,
                              void* d_out, int out_size, void* d_ws, size_t ws_size,
                              hipStream_t stream) {
  (void)in_sizes; (void)n_in; (void)out_size; (void)ws_size;

  const float* x_enc  = (const float*)d_in[0];
  const float* e_Wih0 = (const float*)d_in[4];
  const float* e_Whh0 = (const float*)d_in[5];
  const float* e_bih0 = (const float*)d_in[6];
  const float* e_bhh0 = (const float*)d_in[7];
  const float* e_Wih1 = (const float*)d_in[8];
  const float* e_Whh1 = (const float*)d_in[9];
  const float* e_bih1 = (const float*)d_in[10];
  const float* e_bhh1 = (const float*)d_in[11];
  const float* d_Wih0 = (const float*)d_in[12];
  const float* d_Whh0 = (const float*)d_in[13];
  const float* d_bih0 = (const float*)d_in[14];
  const float* d_bhh0 = (const float*)d_in[15];
  const float* d_Wih1 = (const float*)d_in[16];
  const float* d_Whh1 = (const float*)d_in[17];
  const float* d_bih1 = (const float*)d_in[18];
  const float* d_bhh1 = (const float*)d_in[19];
  const float* fc_W   = (const float*)d_in[20];
  const float* fc_b   = (const float*)d_in[21];
  float* out = (float*)d_out;

  // ---- workspace carve (bump allocator, 256B aligned) ----
  uint8_t* ws = (uint8_t*)d_ws;
  size_t off = 0;
  auto alloc = [&](size_t bytes) -> void* {
    void* p = ws + off;
    off = (off + bytes + 255) & ~(size_t)255;
    return p;
  };
  const size_t IH_E_HALFS = (size_t)128 * KT_E * 512;    // 720,896
  const size_t HH_HALFS   = (size_t)128 * KT_H * 512;    // 1,048,576
  const size_t FC_HALFS   = (size_t)NT_FC * KT_H * 512;  // 172,032

  unsigned short* we0ih = (unsigned short*)alloc(IH_E_HALFS * 2);
  unsigned short* we0hh = (unsigned short*)alloc(HH_HALFS * 2);
  unsigned short* we1ih = (unsigned short*)alloc(HH_HALFS * 2);
  unsigned short* we1hh = (unsigned short*)alloc(HH_HALFS * 2);
  unsigned short* wd0ih = (unsigned short*)alloc(IH_E_HALFS * 2);
  unsigned short* wd0hh = (unsigned short*)alloc(HH_HALFS * 2);
  unsigned short* wd1ih = (unsigned short*)alloc(HH_HALFS * 2);
  unsigned short* wd1hh = (unsigned short*)alloc(HH_HALFS * 2);
  unsigned short* wfc   = (unsigned short*)alloc(FC_HALFS * 2);
  float* b_e0 = (float*)alloc(G4H * 4);
  float* b_e1 = (float*)alloc(G4H * 4);
  float* b_d0 = (float*)alloc(G4H * 4);
  float* b_d1 = (float*)alloc(G4H * 4);
  float* gates = (float*)alloc((size_t)Bd * G4H * 4);          // 4 MB
  unsigned short* Xstage = (unsigned short*)alloc((size_t)Bd * KPAD_E * 2);
  unsigned short* h0 = (unsigned short*)alloc((size_t)Bd * Hd * 2);
  unsigned short* h1 = (unsigned short*)alloc((size_t)Bd * Hd * 2);
  float* c0 = (float*)alloc((size_t)Bd * Hd * 4);
  float* c1 = (float*)alloc((size_t)Bd * Hd * 4);

  // ---- one-time-per-call state init (graph-safe async memsets) ----
  hipMemsetAsync(h0, 0, (size_t)Bd * Hd * 2, stream);
  hipMemsetAsync(h1, 0, (size_t)Bd * Hd * 2, stream);
  hipMemsetAsync(c0, 0, (size_t)Bd * Hd * 4, stream);
  hipMemsetAsync(c1, 0, (size_t)Bd * Hd * 4, stream);

  // ---- weight / bias packing ----
  const int TB = 256;
  auto packW = [&](const float* W, unsigned short* Wp, int Nw, int Kw, int kT) {
    int nT = (Nw + 15) / 16;
    int total = nT * kT * 512;
    pack_w_kernel<<<(total + TB - 1) / TB, TB, 0, stream>>>(W, Wp, Nw, Kw, kT, total);
  };
  packW(e_Wih0, we0ih, G4H, Ed, KT_E);
  packW(e_Whh0, we0hh, G4H, Hd, KT_H);
  packW(e_Wih1, we1ih, G4H, Hd, KT_H);
  packW(e_Whh1, we1hh, G4H, Hd, KT_H);
  packW(d_Wih0, wd0ih, G4H, Ed, KT_E);
  packW(d_Whh0, wd0hh, G4H, Hd, KT_H);
  packW(d_Wih1, wd1ih, G4H, Hd, KT_H);
  packW(d_Whh1, wd1hh, G4H, Hd, KT_H);
  packW(fc_W,   wfc,   Cd,  Hd, KT_H);

  bias_sum_kernel<<<G4H / TB, TB, 0, stream>>>(e_bih0, e_bhh0, b_e0);
  bias_sum_kernel<<<G4H / TB, TB, 0, stream>>>(e_bih1, e_bhh1, b_e1);
  bias_sum_kernel<<<G4H / TB, TB, 0, stream>>>(d_bih0, d_bhh0, b_d0);
  bias_sum_kernel<<<G4H / TB, TB, 0, stream>>>(d_bih1, d_bhh1, b_d1);

  const int gemmBlocks = (MT2 * NT2) / 8;            // 128 (8 waves/block)
  const int cellBlocks = (Bd * Hd) / TB;             // 1024
  const int convBlocks = (Bd * KPAD_E + TB - 1) / TB;
  const int fcBlocks   = (MT_B * NT_FC + 7) / 8;     // 84

  // ---- encoder: interleave layer0/layer1 per timestep (no ys buffer) ----
  for (int t = 0; t < Td; ++t) {
    conv_x_kernel<<<convBlocks, TB, 0, stream>>>(x_enc, Xstage, t);
    gemm_gates_kernel<<<gemmBlocks, TB, 0, stream>>>(
        Xstage, KT_E, we0ih, h0, KT_H, we0hh, b_e0, gates);
    cell_kernel<<<cellBlocks, TB, 0, stream>>>(gates, c0, h0);
    gemm_gates_kernel<<<gemmBlocks, TB, 0, stream>>>(
        h0, KT_H, we1ih, h1, KT_H, we1hh, b_e1, gates);
    cell_kernel<<<cellBlocks, TB, 0, stream>>>(gates, c1, h1);
  }
  // Xstage now holds x_enc[:, T-1, :] == decoder's inp0 (pad region is zero).

  // ---- decoder: autoregressive, out feeds back as next input ----
  for (int t = 0; t < Pd; ++t) {
    gemm_gates_kernel<<<gemmBlocks, TB, 0, stream>>>(
        Xstage, KT_E, wd0ih, h0, KT_H, wd0hh, b_d0, gates);
    cell_kernel<<<cellBlocks, TB, 0, stream>>>(gates, c0, h0);
    gemm_gates_kernel<<<gemmBlocks, TB, 0, stream>>>(
        h0, KT_H, wd1ih, h1, KT_H, wd1hh, b_d1, gates);
    cell_kernel<<<cellBlocks, TB, 0, stream>>>(gates, c1, h1);
    fc_kernel<<<fcBlocks, TB, 0, stream>>>(h1, wfc, fc_b, out, Xstage, t);
  }
}
